// SwinTransformerQA_74921409511853
// MI455X (gfx1250) — matrix operand order repaired and checked by reading, not compiled, not run on hardware
//
#include <hip/hip_runtime.h>
#include <hip/hip_bf16.h>

typedef __attribute__((ext_vector_type(16))) _Float16 v16h;
typedef __attribute__((ext_vector_type(8)))  _Float16 v8h;
typedef __attribute__((ext_vector_type(8)))  float    v8f;
typedef __attribute__((ext_vector_type(4)))  unsigned int u32x4;
typedef __attribute__((ext_vector_type(8)))  int          i32x8;
typedef __attribute__((ext_vector_type(4)))  int          i32x4;

#if defined(__has_builtin)
#if __has_builtin(__builtin_amdgcn_tensor_load_to_lds) && \
    __has_builtin(__builtin_amdgcn_s_wait_tensorcnt)
#define USE_TDM 1
#endif
#endif

static __device__ __forceinline__ v16h cat8(v8h lo, v8h hi) {
    return __builtin_shufflevector(lo, hi, 0,1,2,3,4,5,6,7,8,9,10,11,12,13,14,15);
}
// A-fragment: halves at p[0..15] (contiguous k order)
static __device__ __forceinline__ v16h ld_afrag(const _Float16* p) {
    v8h lo = *(const v8h*)p;
    v8h hi = *(const v8h*)(p + 8);
    return cat8(lo, hi);
}
static __device__ __forceinline__ v16h ld_cfrag(const _Float16* p) {
    return *(const v16h*)p;
}

#ifdef USE_TDM
// Issue a 4 KB contiguous global->LDS DMA via the Tensor Data Mover.
// D# per CDNA5 ISA ch.8: type=2, count=1, data_size=8B, 1-D tile of 512 elems.
static __device__ __forceinline__ void tdm_load_4k(const _Float16* gsrc, _Float16* ldst)
{
    unsigned long long ga = (unsigned long long)(uintptr_t)gsrc;
    unsigned int       la = (unsigned int)(uintptr_t)ldst;   // LDS byte offset
    u32x4 g0;
    g0[0] = 1u;                                              // count=1
    g0[1] = la;                                              // lds_addr
    g0[2] = (unsigned int)ga;                                // global_addr[31:0]
    g0[3] = (unsigned int)((ga >> 32) & 0x01FFFFFFu) | (2u << 30);  // addr hi | type=2
    const unsigned int nd = 512;                             // 4096 B / 8 B
    i32x8 g1;
    g1[0] = (3 << 16);                                       // data_size = 8B
    g1[1] = (int)((nd & 0xFFFFu) << 16);                     // tensor_dim0 lo16 @ [63:48]
    g1[2] = (int)((nd >> 16) | (1u << 16));                  // dim0 hi16 | tensor_dim1=1
    g1[3] = (int)(nd << 16);                                 // tile_dim0 @ [127:112]
    g1[4] = 0;                                               // tile_dim1/2 unused
    g1[5] = (int)nd;                                         // tensor_dim0_stride
    g1[6] = 0;
    g1[7] = 0;
    i32x4 z4 = {0, 0, 0, 0};
    i32x8 z8 = {0, 0, 0, 0, 0, 0, 0, 0};
    __builtin_amdgcn_tensor_load_to_lds(g0, g1, z4, z4, z8, 0);
}
#endif

// ---------------------------------------------------------------------------
// WMMA GEMM, all-f16 data, f32 accumulate:
//   C[M,N] = act(A[M,K] * W + bias[N]) + res[M,N]
// W is pre-packed fragment-major: stream of 4 KB images per (nblk, kstep).
// B staging: TDM async DMA (double buffered) or per-lane vector copy fallback.
// act: 0=none, 1=GELU(exact), 2=ReLU.  Output f16 (C16) or f32 (C32).
// ---------------------------------------------------------------------------
#define GBM 64
#define GBN 64
#define GBK 32

__global__ __launch_bounds__(128)
void gemm_wmma_kernel(const _Float16* __restrict__ A, const _Float16* __restrict__ Bp,
                      const float* __restrict__ bias, const _Float16* __restrict__ res,
                      _Float16* __restrict__ C16, float* __restrict__ C32,
                      int M, int N, int K, int act)
{
    __shared__ __align__(32) _Float16 As[GBM][GBK];      // 4 KB row-major
    __shared__ __align__(32) _Float16 BsF[2][GBK * GBN]; // 2 x 4 KB fragment-major

    const int nb   = blockIdx.x;
    const int mblk = blockIdx.y * GBM;
    const int nblk = nb * GBN;
    const int tid  = threadIdx.x;
    const int wave = tid >> 5;
    const int lane = tid & 31;
    const int l    = lane & 15;
    const int hi   = lane >> 4;
    const int S    = (K + 31) >> 5;

    // hoisted A staging coords: 2 groups of 8 halves per thread
    int ar[2], ac[2];
    bool am[2];
    const _Float16* ap[2];
    #pragma unroll
    for (int g = 0; g < 2; ++g) {
        int i = tid * 8 + g * 1024;          // 2048 halves / (128 thr * 8)
        ar[g] = i >> 5; ac[g] = i & 31;      // ac multiple of 8
        int gm = mblk + ar[g];
        am[g] = gm < M;
        ap[g] = A + (size_t)gm * K;
    }

    const v8f vzero = {0.f,0.f,0.f,0.f,0.f,0.f,0.f,0.f};
    v8f acc[4];
    #pragma unroll
    for (int i = 0; i < 4; ++i) acc[i] = vzero;

    auto loadB = [&](int s, int buf) {
        const _Float16* src = Bp + ((size_t)nb * S + s) * 2048;
#ifdef USE_TDM
        if (wave == 0) tdm_load_4k(src, &BsF[buf][0]);
#else
        const v8h* g = (const v8h*)src;
        *(v8h*)&BsF[buf][tid * 8]        = g[tid];
        *(v8h*)&BsF[buf][1024 + tid * 8] = g[tid + 128];
#endif
    };

    loadB(0, 0);
    for (int s = 0; s < S; ++s) {
        const int buf = s & 1;
        const int k0  = s * GBK;
        // ---- stage A tile (pure f16 copy, group-level bounds; K is always 8-aligned)
        #pragma unroll
        for (int g = 0; g < 2; ++g) {
            v8h x = { (_Float16)0, (_Float16)0, (_Float16)0, (_Float16)0,
                      (_Float16)0, (_Float16)0, (_Float16)0, (_Float16)0 };
            int gk = k0 + ac[g];
            if (am[g] && gk + 8 <= K) x = *(const v8h*)(ap[g] + gk);
            *(v8h*)&As[ar[g]][ac[g]] = x;
        }
        // ---- prefetch next B image / wait current
        if (s + 1 < S) {
            loadB(s + 1, buf ^ 1);
#ifdef USE_TDM
            if (wave == 0) __builtin_amdgcn_s_wait_tensorcnt((short)1);
#endif
        } else {
#ifdef USE_TDM
            if (wave == 0) __builtin_amdgcn_s_wait_tensorcnt((short)0);
#endif
        }
        __syncthreads();
        v16h a = ld_afrag(&As[wave * 16 + l][hi * 16]);
        #pragma unroll
        for (int nt = 0; nt < 4; ++nt) {
            v16h bf = ld_cfrag(&BsF[buf][((nt * 2 + hi) * 16 + l) * 16]);
            acc[nt] = __builtin_amdgcn_wmma_f32_16x16x32_f16(
                false, a, false, bf, (short)0, acc[nt], false, false);
        }
        __syncthreads();
    }

    // ---- epilogue ----
    #pragma unroll
    for (int nt = 0; nt < 4; ++nt) {
        int gn = nblk + nt * 16 + l;
        if (gn >= N) continue;
        float bv = bias ? bias[gn] : 0.f;
        #pragma unroll
        for (int r = 0; r < 8; ++r) {
            int gm = mblk + wave * 16 + (hi ? r + 8 : r);
            if (gm >= M) continue;
            float v = acc[nt][r] + bv;
            if (act == 1)      v = 0.5f * v * (1.f + erff(v * 0.70710678118f));
            else if (act == 2) v = fmaxf(v, 0.f);
            size_t o = (size_t)gm * N + gn;
            if (res) v += (float)res[o];
            if (C32) C32[o] = v;
            else     C16[o] = (_Float16)v;
        }
    }
}

// ---------------------------------------------------------------------------
// Weight pre-pack: f32 [K,N] row-major -> f16 fragment-major K-step streams.
// dst[((nb*S + s)*2048) + ((nt*2+hb)*16+lc)*16 + ii] = W[s*32+hb*16+ii][nb*64+nt*16+lc]
// ---------------------------------------------------------------------------
__global__ void pack_weight(const float* __restrict__ Wsrc, _Float16* __restrict__ dst,
                            int K, int N)
{
    int S  = (K + 31) >> 5;
    int NB = (N + 63) >> 6;
    size_t total = (size_t)NB * S * 2048;
    size_t i = (size_t)blockIdx.x * blockDim.x + threadIdx.x;
    if (i >= total) return;
    int off = (int)(i & 2047);
    size_t blk = i >> 11;
    int s  = (int)(blk % S);
    int nb = (int)(blk / S);
    int ii = off & 15;
    int lc = (off >> 4) & 15;
    int hb = (off >> 8) & 1;
    int nt = off >> 9;
    int n = nb * 64 + nt * 16 + lc;
    int k = s * 32 + hb * 16 + ii;
    float v = (n < N && k < K) ? Wsrc[(size_t)k * N + n] : 0.f;
    dst[i] = (_Float16)v;
}

// ---------------------------------------------------------------------------
// Fused window attention (f16 I/O): one block (128 thr) per (window, head).
// ---------------------------------------------------------------------------
__global__ __launch_bounds__(128)
void window_attn_kernel(const _Float16* __restrict__ qkv, const float* __restrict__ rpb,
                        _Float16* __restrict__ out, int Bn, int H, int W, int C,
                        int heads, int shift)
{
    const int NW = 49;
    const int HD = 32;
    __shared__ __align__(32) _Float16 Qs[64][32];
    __shared__ __align__(32) _Float16 Ks[64][32];
    __shared__ __align__(32) _Float16 VsT[32][64];
    __shared__ __align__(32) _Float16 Ps[4][16][64];

    const int nWx  = W / 7;
    const int nWy  = H / 7;
    const int widx = blockIdx.x;
    const int head = blockIdx.y;
    const int b    = widx / (nWx * nWy);
    const int wrem = widx % (nWx * nWy);
    const int wh   = wrem / nWx;
    const int ww   = wrem % nWx;
    const int C3   = 3 * C;

    const int tid  = threadIdx.x;
    const int wave = tid >> 5;
    const int lane = tid & 31;
    const int l    = lane & 15;
    const int hi   = lane >> 4;
    const float qscale = 0.17677669529f;   // 1/sqrt(32)

    for (int i = tid; i < 64 * HD; i += 128) {
        int n = i >> 5, d = i & 31;
        float qv = 0.f, kv = 0.f, vv = 0.f;
        if (n < NW) {
            int ih = n / 7, iw = n % 7;
            int sh = (wh * 7 + ih + shift) % H;
            int sw = (ww * 7 + iw + shift) % W;
            size_t tok = ((size_t)b * H + sh) * W + sw;
            const _Float16* base = qkv + tok * C3 + head * HD + d;
            qv = (float)base[0]; kv = (float)base[C]; vv = (float)base[2 * C];
        }
        Qs[n][d]  = (_Float16)(qv * qscale);
        Ks[n][d]  = (_Float16)kv;
        VsT[d][n] = (_Float16)vv;
    }
    __syncthreads();

    v16h a = ld_afrag(&Qs[wave * 16 + l][hi * 16]);
    const v8f vzero = {0.f,0.f,0.f,0.f,0.f,0.f,0.f,0.f};
    float lg[4][8];
    #pragma unroll
    for (int nt = 0; nt < 4; ++nt) {
        v16h bf = ld_cfrag(&Ks[nt * 16 + l][hi * 16]);
        v8f s = __builtin_amdgcn_wmma_f32_16x16x32_f16(
            false, a, false, bf, (short)0, vzero, false, false);
        #pragma unroll
        for (int r = 0; r < 8; ++r) lg[nt][r] = s[r];
    }

    #pragma unroll
    for (int nt = 0; nt < 4; ++nt) {
        int col = nt * 16 + l;
        #pragma unroll
        for (int r = 0; r < 8; ++r) {
            int row = wave * 16 + (hi ? r + 8 : r);
            float v = lg[nt][r];
            if (col >= NW || row >= NW) {
                v = -1e30f;
            } else {
                int dy = row / 7 - col / 7 + 6;
                int dx = row % 7 - col % 7 + 6;
                v += rpb[(dy * 13 + dx) * heads + head];
                if (shift > 0) {
                    int ghr = wh * 7 + row / 7, gwr = ww * 7 + row % 7;
                    int ghc = wh * 7 + col / 7, gwc = ww * 7 + col % 7;
                    int rr = (ghr < H - 7 ? 0 : (ghr < H - 3 ? 1 : 2)) * 3
                           + (gwr < W - 7 ? 0 : (gwr < W - 3 ? 1 : 2));
                    int rc = (ghc < H - 7 ? 0 : (ghc < H - 3 ? 1 : 2)) * 3
                           + (gwc < W - 7 ? 0 : (gwc < W - 3 ? 1 : 2));
                    if (rr != rc) v -= 100.f;
                }
            }
            lg[nt][r] = v;
        }
    }

    #pragma unroll
    for (int r = 0; r < 8; ++r) {
        float m = -1e30f;
        #pragma unroll
        for (int nt = 0; nt < 4; ++nt) m = fmaxf(m, lg[nt][r]);
        for (int off = 1; off < 16; off <<= 1) m = fmaxf(m, __shfl_xor(m, off, 32));
        float e[4], sum = 0.f;
        #pragma unroll
        for (int nt = 0; nt < 4; ++nt) { e[nt] = __expf(lg[nt][r] - m); sum += e[nt]; }
        for (int off = 1; off < 16; off <<= 1) sum += __shfl_xor(sum, off, 32);
        float inv = 1.f / sum;
        int row = hi ? r + 8 : r;
        #pragma unroll
        for (int nt = 0; nt < 4; ++nt)
            Ps[wave][row][nt * 16 + l] = (_Float16)(e[nt] * inv);
    }
    __syncthreads();

    v8f acc2[2];
    acc2[0] = vzero; acc2[1] = vzero;
    #pragma unroll
    for (int kk = 0; kk < 2; ++kk) {
        v16h a2 = ld_afrag(&Ps[wave][l][kk * 32 + hi * 16]);
        #pragma unroll
        for (int nt = 0; nt < 2; ++nt) {
            v16h bf = ld_cfrag(&VsT[nt * 16 + l][kk * 32 + hi * 16]);
            acc2[nt] = __builtin_amdgcn_wmma_f32_16x16x32_f16(
                false, a2, false, bf, (short)0, acc2[nt], false, false);
        }
    }

    #pragma unroll
    for (int nt = 0; nt < 2; ++nt) {
        int d = nt * 16 + l;
        #pragma unroll
        for (int r = 0; r < 8; ++r) {
            int row = wave * 16 + (hi ? r + 8 : r);
            if (row < NW) {
                int ih = row / 7, iw = row % 7;
                int sh = (wh * 7 + ih + shift) % H;
                int sw = (ww * 7 + iw + shift) % W;
                size_t tok = ((size_t)b * H + sh) * W + sw;
                out[tok * C + head * HD + d] = (_Float16)acc2[nt][r];
            }
        }
    }
}

// ---------------------------------------------------------------------------
// LayerNorm (f16 in/out, f32 stats): one block (128 thr) per token.
// ---------------------------------------------------------------------------
__global__ __launch_bounds__(128)
void ln_kernel(const _Float16* __restrict__ in, const float* __restrict__ g,
               const float* __restrict__ b, _Float16* __restrict__ out, int C)
{
    int tok = blockIdx.x;
    const _Float16* x = in + (size_t)tok * C;
    _Float16* o = out + (size_t)tok * C;
    float s = 0.f, s2 = 0.f;
    for (int i = threadIdx.x; i < C; i += blockDim.x) {
        float v = (float)x[i]; s += v; s2 += v * v;
    }
    for (int off = 16; off > 0; off >>= 1) {
        s  += __shfl_down(s, off, 32);
        s2 += __shfl_down(s2, off, 32);
    }
    __shared__ float ls[4], ls2[4];
    int wave = threadIdx.x >> 5, lane = threadIdx.x & 31;
    if (lane == 0) { ls[wave] = s; ls2[wave] = s2; }
    __syncthreads();
    if (threadIdx.x == 0) {
        float t = 0.f, t2 = 0.f;
        for (int i = 0; i < 4; ++i) { t += ls[i]; t2 += ls2[i]; }
        ls[0] = t; ls2[0] = t2;
    }
    __syncthreads();
    float mu  = ls[0] / C;
    float var = ls2[0] / C - mu * mu;
    float inv = rsqrtf(var + 1e-5f);
    for (int i = threadIdx.x; i < C; i += blockDim.x)
        o[i] = (_Float16)(((float)x[i] - mu) * inv * g[i] + b[i]);
}

// ---------------------------------------------------------------------------
// Patch-embed im2col: images f32 [B,3,224,224] -> f16 [B*56*56, 48]
// ---------------------------------------------------------------------------
__global__ void im2col_patch(const float* __restrict__ img, _Float16* __restrict__ out, int Bn)
{
    size_t total = (size_t)Bn * 56 * 56 * 48;
    size_t idx = (size_t)blockIdx.x * blockDim.x + threadIdx.x;
    if (idx >= total) return;
    int j = (int)(idx % 48); size_t t = idx / 48;
    int pw = (int)(t % 56); t /= 56;
    int ph = (int)(t % 56); int b = (int)(t / 56);
    int ch = j % 3; int pj = j / 3;
    int jj = pj % 4; int ii = pj / 4;
    out[idx] = (_Float16)img[(((size_t)b * 3 + ch) * 224 + (ph * 4 + ii)) * 224 + (pw * 4 + jj)];
}

// ---------------------------------------------------------------------------
// Patch-merge gather (f16): x [B,H,W,C] -> out [B,H/2,W/2,4C]
// ---------------------------------------------------------------------------
__global__ void merge_gather(const _Float16* __restrict__ x, _Float16* __restrict__ out,
                             int Bn, int H, int W, int C)
{
    size_t total = (size_t)Bn * (H / 2) * (W / 2) * 4 * C;
    size_t i = (size_t)blockIdx.x * blockDim.x + threadIdx.x;
    if (i >= total) return;
    int C4 = 4 * C;
    int cc = (int)(i % C4); size_t t = i / C4;
    int w2 = (int)(t % (W / 2)); t /= (W / 2);
    int h2 = (int)(t % (H / 2)); int b = (int)(t / (H / 2));
    int grp = cc / C, c = cc % C;
    int dh = (grp == 1 || grp == 3) ? 1 : 0;
    int dw = (grp >= 2) ? 1 : 0;
    out[i] = x[(((size_t)b * H + (2 * h2 + dh)) * W + (2 * w2 + dw)) * C + c];
}

// ---------------------------------------------------------------------------
// Global average pool (f16 in/out): x [B,T,C] -> feat [B,C]
// ---------------------------------------------------------------------------
__global__ void pool_kernel(const _Float16* __restrict__ x, _Float16* __restrict__ feat,
                            int T, int C)
{
    int b = blockIdx.x;
    for (int c = threadIdx.x; c < C; c += blockDim.x) {
        float s = 0.f;
        for (int t = 0; t < T; ++t) s += (float)x[((size_t)b * T + t) * C + c];
        feat[(size_t)b * C + c] = (_Float16)(s / (float)T);
    }
}

// ---------------------------------------------------------------------------
// Host orchestration
// ---------------------------------------------------------------------------
extern "C" void kernel_launch(void* const* d_in, const int* in_sizes, int n_in,
                              void* d_out, int out_size, void* d_ws, size_t ws_size,
                              hipStream_t stream)
{
    (void)in_sizes; (void)n_in; (void)out_size; (void)ws_size;
    const float* const* in = (const float* const*)d_in;
    int idx = 0;
    const float* images = in[idx++];
    const float* pe_w  = in[idx++];
    const float* pe_b  = in[idx++];
    const float* pe_ng = in[idx++];
    const float* pe_nb = in[idx++];

    struct Blk { const float *n1g,*n1b,*qkvw,*qkvb,*rpb,*projw,*projb,
                             *n2g,*n2b,*fc1w,*fc1b,*fc2w,*fc2b; };
    struct Mrg { const float *ng,*nb,*rw; };
    static const int depths[4] = {2, 2, 6, 2};
    static const int headsA[4] = {3, 6, 12, 24};
    Blk blk[4][6]; Mrg mrg[3];
    for (int s = 0; s < 4; ++s) {
        for (int j = 0; j < depths[s]; ++j) {
            Blk& p = blk[s][j];
            p.n1g  = in[idx++]; p.n1b  = in[idx++];
            p.qkvw = in[idx++]; p.qkvb = in[idx++];
            p.rpb  = in[idx++];
            p.projw= in[idx++]; p.projb= in[idx++];
            p.n2g  = in[idx++]; p.n2b  = in[idx++];
            p.fc1w = in[idx++]; p.fc1b = in[idx++];
            p.fc2w = in[idx++]; p.fc2b = in[idx++];
        }
        if (s < 3) { mrg[s].ng = in[idx++]; mrg[s].nb = in[idx++]; mrg[s].rw = in[idx++]; }
    }
    const float* fng = in[idx++]; const float* fnb = in[idx++];
    const float* h1w = in[idx++]; const float* h1b = in[idx++];
    const float* h2w = in[idx++]; const float* h2b = in[idx++];

    const int Bn = 32;
    const size_t T0 = (size_t)Bn * 56 * 56;             // 100352 tokens
    auto alignup = [](size_t x) { return (x + 255) & ~(size_t)255; };
    char* ws = (char*)d_ws;
    _Float16* bufX = (_Float16*)ws; ws += alignup(T0 * 96  * 2);
    _Float16* bufY = (_Float16*)ws; ws += alignup(T0 * 96  * 2);
    _Float16* bufQ = (_Float16*)ws; ws += alignup(T0 * 288 * 2);
    _Float16* bufH = (_Float16*)ws; ws += alignup(T0 * 384 * 2);
    _Float16* bufT = (_Float16*)ws; ws += alignup(T0 * 96  * 2);

    // ---- packed weight region: carve + pack (deterministic every call) ----
    auto pksize = [](int K, int N) {
        return (size_t)((N + 63) / 64) * ((K + 31) / 32) * 2048;   // halves
    };
    auto pack = [&](const float* src, int K, int N) -> _Float16* {
        _Float16* dst = (_Float16*)ws;
        size_t total = pksize(K, N);
        ws += alignup(total * 2);
        pack_weight<<<(unsigned)((total + 255) / 256), 256, 0, stream>>>(src, dst, K, N);
        return dst;
    };

    _Float16* pk_pe = pack(pe_w, 48, 96);
    struct BlkP { _Float16 *qkv, *proj, *fc1, *fc2; };
    BlkP pblk[4][6];
    _Float16* pmrg[3];
    {
        int C = 96;
        for (int s = 0; s < 4; ++s) {
            for (int j = 0; j < depths[s]; ++j) {
                pblk[s][j].qkv  = pack(blk[s][j].qkvw, C, 3 * C);
                pblk[s][j].proj = pack(blk[s][j].projw, C, C);
                pblk[s][j].fc1  = pack(blk[s][j].fc1w, C, 4 * C);
                pblk[s][j].fc2  = pack(blk[s][j].fc2w, 4 * C, C);
            }
            if (s < 3) { pmrg[s] = pack(mrg[s].rw, 4 * C, 2 * C); C *= 2; }
        }
    }
    _Float16* ph1 = pack(h1w, 768, 256);
    _Float16* ph2 = pack(h2w, 256, 1);

    auto gemm = [&](const _Float16* A, const _Float16* Bp, const float* bias,
                    const _Float16* res, _Float16* C16, float* C32,
                    int M, int N, int K, int act) {
        dim3 g((N + GBN - 1) / GBN, (M + GBM - 1) / GBM);
        gemm_wmma_kernel<<<g, 128, 0, stream>>>(A, Bp, bias, res, C16, C32, M, N, K, act);
    };

    // ---- patch embed
    {
        size_t tot = T0 * 48;
        im2col_patch<<<(unsigned)((tot + 255) / 256), 256, 0, stream>>>(images, bufH, Bn);
        gemm(bufH, pk_pe, pe_b, nullptr, bufY, nullptr, (int)T0, 96, 48, 0);
        ln_kernel<<<(unsigned)T0, 128, 0, stream>>>(bufY, pe_ng, pe_nb, bufX, 96);
    }

    int H = 56, W = 56, C = 96;
    for (int s = 0; s < 4; ++s) {
        int T = Bn * H * W;
        int nH = headsA[s];
        for (int j = 0; j < depths[s]; ++j) {
            const Blk& p = blk[s][j];
            const BlkP& q = pblk[s][j];
            int shift = (j % 2 == 1 && H > 7) ? 3 : 0;
            ln_kernel<<<(unsigned)T, 128, 0, stream>>>(bufX, p.n1g, p.n1b, bufY, C);
            gemm(bufY, q.qkv, p.qkvb, nullptr, bufQ, nullptr, T, 3 * C, C, 0);
            dim3 ag((unsigned)(Bn * (H / 7) * (W / 7)), (unsigned)nH);
            window_attn_kernel<<<ag, 128, 0, stream>>>(bufQ, p.rpb, bufT, Bn, H, W, C, nH, shift);
            gemm(bufT, q.proj, p.projb, bufX, bufX, nullptr, T, C, C, 0);
            ln_kernel<<<(unsigned)T, 128, 0, stream>>>(bufX, p.n2g, p.n2b, bufY, C);
            gemm(bufY, q.fc1, p.fc1b, nullptr, bufH, nullptr, T, 4 * C, C, 1);
            gemm(bufH, q.fc2, p.fc2b, bufX, bufX, nullptr, T, C, 4 * C, 0);
        }
        if (s < 3) {
            int T2 = Bn * (H / 2) * (W / 2);
            size_t tot2 = (size_t)T2 * 4 * C;
            merge_gather<<<(unsigned)((tot2 + 255) / 256), 256, 0, stream>>>(bufX, bufY, Bn, H, W, C);
            ln_kernel<<<(unsigned)T2, 128, 0, stream>>>(bufY, mrg[s].ng, mrg[s].nb, bufH, 4 * C);
            gemm(bufH, pmrg[s], nullptr, nullptr, bufX, nullptr, T2, 2 * C, 4 * C, 0);
            H /= 2; W /= 2; C *= 2;
        }
    }

    // ---- final norm, pool, head
    int T = Bn * H * W;   // 32 * 49
    ln_kernel<<<(unsigned)T, 128, 0, stream>>>(bufX, fng, fnb, bufY, C);
    pool_kernel<<<Bn, 256, 0, stream>>>(bufY, bufT, H * W, C);
    gemm(bufT, ph1, h1b, nullptr, bufH, nullptr, Bn, 256, C, 2);
    gemm(bufH, ph2, h2b, nullptr, nullptr, (float*)d_out, Bn, 1, 256, 0);
}
